// HeteroRouter_23553600651626
// MI455X (gfx1250) — matrix operand order repaired and checked by reading, not compiled, not run on hardware
//
#include <hip/hip_runtime.h>
#include <hip/hip_bf16.h>

typedef __bf16 bf16_t;
typedef __attribute__((ext_vector_type(16))) __bf16 v16bf;
typedef __attribute__((ext_vector_type(8)))  float  v8f;

#define N_TOKENS 131072
#define EMBED    1024
#define NEXP     64
#define BETA     0.1f
#define WSTRIDE  1032   // 1024 + 8 pad bf16 -> row stride 516 dwords == 4 (mod 64): conflict-minimal

// ---------------- prep: W fp32 -> bf16 in workspace ----------------
__global__ void w_to_bf16_kernel(const float* __restrict__ W, bf16_t* __restrict__ Wb) {
    int i = (blockIdx.x * blockDim.x + threadIdx.x) * 4;
    float4 f = *(const float4*)(W + i);
    Wb[i + 0] = (bf16_t)f.x;
    Wb[i + 1] = (bf16_t)f.y;
    Wb[i + 2] = (bf16_t)f.z;
    Wb[i + 3] = (bf16_t)f.w;
}

// ---------------- helpers ----------------
__device__ __forceinline__ bool tk_better(float a, int ia, float b, int ib) {
    return (a > b) || (a == b && ia < ib);   // jax top_k tie-break: lower index wins
}
__device__ __forceinline__ void tk_insert(float v, int i,
                                          float& v1, int& i1, float& v2, int& i2) {
    if (tk_better(v, i, v1, i1)) { v2 = v1; i2 = i1; v1 = v; i1 = i; }
    else if (tk_better(v, i, v2, i2)) { v2 = v; i2 = i; }
}
__device__ __forceinline__ v16bf lds_loadB(const bf16_t* p) {
    union { uint4 u[2]; v16bf v; } t;
    t.u[0] = *(const uint4*)(p);       // 16B aligned: row*2064 + 64k + 16h + {0,32}
    t.u[1] = *(const uint4*)(p + 16);
    return t.v;
}

// ---------------- main fused router kernel ----------------
// 256 threads = 8 waves; each wave = 16 tokens x 64 experts; block = 128 tokens.
__launch_bounds__(256)
__global__ void router_kernel(const float* __restrict__ x,
                              const bf16_t* __restrict__ Wb,
                              const float* __restrict__ costs,
                              float* __restrict__ out) {
    extern __shared__ bf16_t Wlds[];   // 64 x WSTRIDE bf16 (~129 KB)

    const int tid = threadIdx.x;

    // Stage W (bf16) into padded LDS: 8192 uint4 chunks, coalesced.
    #pragma unroll
    for (int it = 0; it < 32; ++it) {
        int c   = it * 256 + tid;        // chunk of 8 bf16
        int row = c >> 7;                // 128 chunks per 1024-elem row
        int col = (c & 127) << 3;
        uint4 v = ((const uint4*)Wb)[c];
        *(uint4*)(&Wlds[row * WSTRIDE + col]) = v;
    }
    __syncthreads();

    const int  lane   = tid & 31;
    const int  wid    = tid >> 5;
    const int  l15    = lane & 15;
    const bool hiHalf = lane >= 16;
    const int  off4   = hiHalf ? 2 : 0;    // float4 offset inside a 32-float K tile
    const int  hi16   = hiHalf ? 8 : 0;    // B-fragment K offset

    const int tokBase = blockIdx.x * 128 + wid * 16;
    const int tA      = tokBase + l15;     // token row this lane feeds into A
    const float4* xr  = (const float4*)(x + (size_t)tA * EMBED);

    const bf16_t* b0p = &Wlds[( 0 + l15) * WSTRIDE + hi16];
    const bf16_t* b1p = &Wlds[(16 + l15) * WSTRIDE + hi16];
    const bf16_t* b2p = &Wlds[(32 + l15) * WSTRIDE + hi16];
    const bf16_t* b3p = &Wlds[(48 + l15) * WSTRIDE + hi16];

    v8f acc0 = {0,0,0,0,0,0,0,0};
    v8f acc1 = {0,0,0,0,0,0,0,0};
    v8f acc2 = {0,0,0,0,0,0,0,0};
    v8f acc3 = {0,0,0,0,0,0,0,0};

    #pragma unroll 2
    for (int k = 0; k < 32; ++k) {
        const int base = k * 8 + off4;
        float4 a0 = xr[base + 0];          // K = 32k+off   .. +3
        float4 a1 = xr[base + 1];          // K = 32k+off+4 .. +7
        float4 a2 = xr[base + 4];          // K = 32k+16+off   .. +3
        float4 a3 = xr[base + 5];          // K = 32k+16+off+4 .. +7
        v16bf A;
        A[0]=(bf16_t)a0.x;  A[1]=(bf16_t)a0.y;  A[2]=(bf16_t)a0.z;  A[3]=(bf16_t)a0.w;
        A[4]=(bf16_t)a1.x;  A[5]=(bf16_t)a1.y;  A[6]=(bf16_t)a1.z;  A[7]=(bf16_t)a1.w;
        A[8]=(bf16_t)a2.x;  A[9]=(bf16_t)a2.y;  A[10]=(bf16_t)a2.z; A[11]=(bf16_t)a2.w;
        A[12]=(bf16_t)a3.x; A[13]=(bf16_t)a3.y; A[14]=(bf16_t)a3.z; A[15]=(bf16_t)a3.w;

        const int kb = k * 32;
        acc0 = __builtin_amdgcn_wmma_f32_16x16x32_bf16(false, A, false, lds_loadB(b0p + kb),
                                                       (short)0, acc0, false, false);
        acc1 = __builtin_amdgcn_wmma_f32_16x16x32_bf16(false, A, false, lds_loadB(b1p + kb),
                                                       (short)0, acc1, false, false);
        acc2 = __builtin_amdgcn_wmma_f32_16x16x32_bf16(false, A, false, lds_loadB(b2p + kb),
                                                       (short)0, acc2, false, false);
        acc3 = __builtin_amdgcn_wmma_f32_16x16x32_bf16(false, A, false, lds_loadB(b3p + kb),
                                                       (short)0, acc3, false, false);
    }

    // per-lane expert bias (-beta * costs[e]) for each column tile
    const float cb0 = -BETA * costs[ 0 + l15];
    const float cb1 = -BETA * costs[16 + l15];
    const float cb2 = -BETA * costs[32 + l15];
    const float cb3 = -BETA * costs[48 + l15];

    float* idxOut = out;                          // [N,2] indices (as float values)
    float* pkOut  = out + (size_t)N_TOKENS * 2;   // [N,2] renormalized top-2 probs
    float* prOut  = out + (size_t)N_TOKENS * 4;   // [N,64] softmax probs

    #pragma unroll
    for (int r = 0; r < 8; ++r) {
        const int token = tokBase + r + (hiHalf ? 8 : 0);

        float l0 = acc0[r] + cb0;
        float l1 = acc1[r] + cb1;
        float l2 = acc2[r] + cb2;
        float l3 = acc3[r] + cb3;

        // softmax over 64 experts: reduce across the 16-lane half-wave
        float m = fmaxf(fmaxf(l0, l1), fmaxf(l2, l3));
        #pragma unroll
        for (int msk = 1; msk < 16; msk <<= 1) m = fmaxf(m, __shfl_xor(m, msk, 32));

        float e0 = __expf(l0 - m), e1 = __expf(l1 - m);
        float e2 = __expf(l2 - m), e3 = __expf(l3 - m);
        float s = e0 + e1 + e2 + e3;
        #pragma unroll
        for (int msk = 1; msk < 16; msk <<= 1) s += __shfl_xor(s, msk, 32);

        const float inv = 1.0f / s;
        float p0 = e0 * inv, p1 = e1 * inv, p2 = e2 * inv, p3 = e3 * inv;

        float* pr = prOut + (size_t)token * 64 + l15;
        pr[0] = p0; pr[16] = p1; pr[32] = p2; pr[48] = p3;

        // per-lane top-2 of its 4 experts, then butterfly-merge across 16 lanes
        float v1 = p0, v2 = p1; int i1 = l15, i2 = 16 + l15;
        if (tk_better(v2, i2, v1, i1)) { float tv = v1; v1 = v2; v2 = tv; int ti = i1; i1 = i2; i2 = ti; }
        tk_insert(p2, 32 + l15, v1, i1, v2, i2);
        tk_insert(p3, 48 + l15, v1, i1, v2, i2);
        #pragma unroll
        for (int msk = 1; msk < 16; msk <<= 1) {
            float w1 = __shfl_xor(v1, msk, 32), w2 = __shfl_xor(v2, msk, 32);
            int   j1 = __shfl_xor(i1, msk, 32), j2 = __shfl_xor(i2, msk, 32);
            tk_insert(w1, j1, v1, i1, v2, i2);
            tk_insert(w2, j2, v1, i1, v2, i2);
        }

        if (l15 == 0) {
            const float tsum = v1 + v2 + 1e-8f;
            idxOut[(size_t)token * 2 + 0] = (float)i1;
            idxOut[(size_t)token * 2 + 1] = (float)i2;
            pkOut [(size_t)token * 2 + 0] = v1 / tsum;
            pkOut [(size_t)token * 2 + 1] = v2 / tsum;
        }
    }
}

extern "C" void kernel_launch(void* const* d_in, const int* in_sizes, int n_in,
                              void* d_out, int out_size, void* d_ws, size_t ws_size,
                              hipStream_t stream) {
    const float* x     = (const float*)d_in[0];   // [131072, 1024] fp32
    const float* W     = (const float*)d_in[1];   // [64, 1024] fp32
    const float* costs = (const float*)d_in[2];   // [64] fp32
    bf16_t* Wb = (bf16_t*)d_ws;                   // 64*1024 bf16 = 128 KB

    // 1) convert W to bf16 (65536 elems, 4 per thread)
    w_to_bf16_kernel<<<64, 256, 0, stream>>>(W, Wb);

    // 2) fused gemm + softmax + top2: 1024 blocks x 256 threads, dynamic LDS for padded W
    const size_t shmem = (size_t)NEXP * WSTRIDE * sizeof(bf16_t);  // 132096 B
    router_kernel<<<N_TOKENS / 128, 256, shmem, stream>>>(x, Wb, costs, (float*)d_out);
}
